// EdgeEarlyInteraction_72421738545175
// MI455X (gfx1250) — compile-verified, matrix operand order checked by reading, not run on hardware
//
#include <hip/hip_runtime.h>
#include <hip/hip_bf16.h>

// ---------------------------------------------------------------------------
// MI455X / gfx1250 implementation: all GEMMs via v_wmma_f32_16x16x32_bf16.
// ---------------------------------------------------------------------------

typedef __bf16 bf16_t;
typedef __attribute__((ext_vector_type(16))) __bf16 v16bf;
typedef __attribute__((ext_vector_type(8)))  __bf16 v8bf;
typedef __attribute__((ext_vector_type(8)))  float  v8f;

// Problem constants (match reference)
#define BATCH      512
#define E_PER_G    48
#define D_FEAT     128
#define NUM_NODES  32768      // 2*B*N
#define NUM_EDGES  49152      // 2*B*E
#define MSG_LD     640        // P*D
#define NS_DIM     64

__device__ __forceinline__ int wave_id() { return (int)(threadIdx.x >> 5); }
__device__ __forceinline__ int lane_id() { return (int)(threadIdx.x & 31); }

// ---- WMMA fragment helpers -------------------------------------------------
// A-operand (16x32 bf16, M x K): lane 0..15 holds row M=lane, K = k0..k0+7 in
// elems 0..7 and K = k0+16..k0+23 in elems 8..15; lanes 16..31 hold the same
// rows shifted by +8 in K (per CDNA5 ISA 7.12.2). B-operand uses the identical
// layout with N in place of M, i.e. load from a row-major N x Kpad (W^T) array.
__device__ __forceinline__ v16bf load_frag(const bf16_t* base, int ld, int k0) {
  int lane = lane_id();
  const bf16_t* p = base + (lane & 15) * ld + k0 + ((lane >> 4) << 3);
  v8bf lo = *(const v8bf*)(p);
  v8bf hi = *(const v8bf*)(p + 16);
  v16bf f;
#pragma unroll
  for (int i = 0; i < 8; ++i) { f[i] = lo[i]; f[i + 8] = hi[i]; }
  return f;
}

// 16x16 f32 tile = sum over K of A(16xK) * B(Kx16), K multiple of 32.
__device__ __forceinline__ v8f tile_gemm(const bf16_t* A, int lda,
                                         const bf16_t* Bt, int ldb, int K) {
  v8f acc = {};
  for (int k0 = 0; k0 < K; k0 += 32) {
    v16bf a = load_frag(A, lda, k0);
    v16bf b = load_frag(Bt, ldb, k0);
    acc = __builtin_amdgcn_wmma_f32_16x16x32_bf16(false, a, false, b,
                                                  (short)0, acc, false, false);
  }
  return acc;
}

// C/D layout: VGPR r -> (M = r + 8*(lane>=16), N = lane&15).
__device__ __forceinline__ void store_lds_act(v8f acc, const float* bias,
                                              int col0, int N, bf16_t* H,
                                              int ldh, bool relu) {
  int lane = lane_id();
  int n = col0 + (lane & 15);
  float bv = (n < N) ? bias[n] : 0.f;   // padded cols -> 0 (weights are 0 too)
  int mo = (lane >> 4) << 3;
#pragma unroll
  for (int r = 0; r < 8; ++r) {
    float v = acc[r] + bv;
    if (relu && v < 0.f) v = 0.f;
    H[(r + mo) * ldh + n] = (bf16_t)v;
  }
}

// ---- weight convert: W (K x N, f32) -> W^T padded (Np x Kp, bf16) ----------
__global__ void k_wcvt(const float* __restrict__ W, bf16_t* __restrict__ Wt,
                       int K, int N, int Kp, int Np) {
  int i = blockIdx.x * blockDim.x + threadIdx.x;
  if (i >= Kp * Np) return;
  int n = i / Kp, k = i % Kp;
  Wt[i] = (bf16_t)((k < K && n < N) ? W[k * N + n] : 0.f);
}

__global__ void k_zero(float* p, size_t n) {
  size_t i = (size_t)blockIdx.x * blockDim.x + threadIdx.x;
  if (i < n) p[i] = 0.f;
}
__global__ void k_copy(float* __restrict__ d, const float* __restrict__ s, size_t n) {
  size_t i = (size_t)blockIdx.x * blockDim.x + threadIdx.x;
  if (i < n) d[i] = s[i];
}

// ---- single GEMM, Kin<=32 (node/edge encoders) -----------------------------
__global__ void k_gemm_k32(const float* __restrict__ X, int ldx, int Kin,
                           const bf16_t* __restrict__ Wt, const float* __restrict__ bias,
                           int N, float* __restrict__ Y, int ldy) {
  __shared__ __align__(16) bf16_t Xs[16 * 32];
  int row0 = blockIdx.x * 16;
  for (int i = threadIdx.x; i < 16 * 32; i += blockDim.x) {
    int r = i >> 5, k = i & 31;
    Xs[i] = (bf16_t)((k < Kin) ? X[(size_t)(row0 + r) * ldx + k] : 0.f);
  }
  __syncthreads();
  for (int ct = wave_id(); ct * 16 < N; ct += (int)(blockDim.x >> 5)) {
    v8f acc = tile_gemm(Xs, 32, Wt + ct * 16 * 32, 32, 32);
    int n = ct * 16 + (lane_id() & 15), mo = (lane_id() >> 4) << 3;
    float bv = bias[n];
#pragma unroll
    for (int r = 0; r < 8; ++r)
      Y[(size_t)(row0 + r + mo) * ldy + n] = acc[r] + bv;
  }
}

// ---- comb = mlp2([enc_e | inter], Wc1, Wc2)  (160 -> 160relu -> 30) --------
__global__ void k_comb(const float* __restrict__ enc_e, const float* __restrict__ interf,
                       int interOff, int useInter,
                       const bf16_t* __restrict__ Wc1t, const float* __restrict__ bc1,
                       const bf16_t* __restrict__ Wc2t, const float* __restrict__ bc2,
                       float* __restrict__ comb) {
  __shared__ __align__(16) bf16_t Xs[16 * 160];
  __shared__ __align__(16) bf16_t Hs[16 * 160];
  int row0 = blockIdx.x * 16;
  for (int i = threadIdx.x; i < 16 * 160; i += blockDim.x) {
    int r = i / 160, k = i % 160;
    int e = row0 + r;
    float v = (k < 32) ? enc_e[(size_t)e * 32 + k]
            : (useInter ? interf[(size_t)e * MSG_LD + interOff + (k - 32)] : 0.f);
    Xs[i] = (bf16_t)v;
  }
  __syncthreads();
  for (int ct = wave_id(); ct < 10; ct += 4) {
    v8f a = tile_gemm(Xs, 160, Wc1t + ct * 16 * 160, 160, 160);
    store_lds_act(a, bc1, ct * 16, 160, Hs, 160, true);
  }
  __syncthreads();
  if (wave_id() < 2) {
    int ct = wave_id();
    v8f a = tile_gemm(Hs, 160, Wc2t + ct * 16 * 160, 160, 160);
    int n = ct * 16 + (lane_id() & 15), mo = (lane_id() >> 4) << 3;
#pragma unroll
    for (int r = 0; r < 8; ++r)
      comb[(size_t)(row0 + r + mo) * 32 + n] = (n < 30) ? a[r] + bc2[n] : 0.f;
  }
}

// ---- fwd/bwd message MLPs + segment-sum scatter ----------------------------
__global__ void k_msg(const float* __restrict__ nodes,
                      const int* __restrict__ from_idx, const int* __restrict__ to_idx,
                      const float* __restrict__ comb,
                      const bf16_t* __restrict__ Wm1t, const float* __restrict__ bm1,
                      const bf16_t* __restrict__ Wm2t, const float* __restrict__ bm2,
                      const bf16_t* __restrict__ Wr1t, const float* __restrict__ br1,
                      const bf16_t* __restrict__ Wr2t, const float* __restrict__ br2,
                      float* __restrict__ msgs, int msgOff, float* __restrict__ agg) {
  __shared__ __align__(16) bf16_t Xf[16 * 160], Xb[16 * 160];
  __shared__ __align__(16) bf16_t Hf[16 * 128], Hb[16 * 128];
  __shared__ int fid[16], gid[16];
  int row0 = blockIdx.x * 16;
  if (threadIdx.x < 16) {
    fid[threadIdx.x] = from_idx[row0 + threadIdx.x];
    gid[threadIdx.x] = to_idx[row0 + threadIdx.x];
  }
  __syncthreads();
  for (int i = threadIdx.x; i < 16 * 160; i += blockDim.x) {
    int r = i / 160, k = i % 160;
    int e = row0 + r;
    float vf, vb;
    if (k < 64)        { vf = nodes[(size_t)fid[r] * 64 + k];        vb = nodes[(size_t)gid[r] * 64 + k]; }
    else if (k < 128)  { vf = nodes[(size_t)gid[r] * 64 + (k - 64)]; vb = nodes[(size_t)fid[r] * 64 + (k - 64)]; }
    else               { float c = comb[(size_t)e * 32 + (k - 128)]; vf = c; vb = c; }
    Xf[i] = (bf16_t)vf; Xb[i] = (bf16_t)vb;
  }
  __syncthreads();
  for (int job = wave_id(); job < 16; job += 4) {
    if (job < 8) {
      v8f a = tile_gemm(Xf, 160, Wm1t + job * 16 * 160, 160, 160);
      store_lds_act(a, bm1, job * 16, 128, Hf, 128, true);
    } else {
      int ct = job - 8;
      v8f a = tile_gemm(Xb, 160, Wr1t + ct * 16 * 160, 160, 160);
      store_lds_act(a, br1, ct * 16, 128, Hb, 128, true);
    }
  }
  __syncthreads();
  for (int ct = wave_id(); ct < 8; ct += 4) {
    v8f af = tile_gemm(Hf, 128, Wm2t + ct * 16 * 128, 128, 128);
    v8f ab = tile_gemm(Hb, 128, Wr2t + ct * 16 * 128, 128, 128);
    int n = ct * 16 + (lane_id() & 15), mo = (lane_id() >> 4) << 3;
    float b2f = bm2[n], b2b = br2[n];
#pragma unroll
    for (int r = 0; r < 8; ++r) {
      int m = r + mo, e = row0 + m;
      float fv = af[r] + b2f, bv = ab[r] + b2b;
      msgs[(size_t)e * MSG_LD + msgOff + n] = fv + bv;
      unsafeAtomicAdd(&agg[(size_t)gid[m] * 128 + n], fv);
      unsafeAtomicAdd(&agg[(size_t)fid[m] * 128 + n], bv);
    }
  }
}

// ---- node update: nodes += mlp2([agg | nodes], Wn1, Wn2) -------------------
__global__ void k_node(float* __restrict__ nodes, const float* __restrict__ agg,
                       const bf16_t* __restrict__ Wn1t, const float* __restrict__ bn1,
                       const bf16_t* __restrict__ Wn2t, const float* __restrict__ bn2) {
  __shared__ __align__(16) bf16_t Xs[16 * 192];
  __shared__ __align__(16) bf16_t Hs[16 * 128];
  int row0 = blockIdx.x * 16;
  for (int i = threadIdx.x; i < 16 * 192; i += blockDim.x) {
    int r = i / 192, k = i % 192;
    int nd = row0 + r;
    float v = (k < 128) ? agg[(size_t)nd * 128 + k] : nodes[(size_t)nd * 64 + (k - 128)];
    Xs[i] = (bf16_t)v;
  }
  __syncthreads();
  for (int ct = wave_id(); ct < 8; ct += 4) {
    v8f a = tile_gemm(Xs, 192, Wn1t + ct * 16 * 192, 192, 192);
    store_lds_act(a, bn1, ct * 16, 128, Hs, 128, true);
  }
  __syncthreads();
  {
    int ct = wave_id();  // 4 waves, 4 col tiles of 64 outputs
    v8f a = tile_gemm(Hs, 128, Wn2t + ct * 16 * 128, 128, 128);
    int n = ct * 16 + (lane_id() & 15), mo = (lane_id() >> 4) << 3;
    float bv = bn2[n];
#pragma unroll
    for (int r = 0; r < 8; ++r)
      nodes[(size_t)(row0 + r + mo) * 64 + n] += a[r] + bv;
  }
}

// ---- tq/tc = mlp2(qf/cf, Wt1, Wt2) * edge_mask  (128 -> 64relu -> 64) ------
__global__ void k_tform(const float* __restrict__ msgs, int parity,
                        const bf16_t* __restrict__ Wt1t, const float* __restrict__ bt1,
                        const bf16_t* __restrict__ Wt2t, const float* __restrict__ bt2,
                        float* __restrict__ out) {
  __shared__ __align__(16) bf16_t Xs[16 * 128];
  __shared__ __align__(16) bf16_t Hs[16 * 64];
  int row0 = blockIdx.x * 16;
  for (int i = threadIdx.x; i < 16 * 128; i += blockDim.x) {
    int r = i >> 7, k = i & 127;
    int row = row0 + r, b = row >> 6, m = row & 63;
    float v = (m < E_PER_G)
                ? msgs[(size_t)((b * 2 + parity) * E_PER_G + m) * MSG_LD + 512 + k] : 0.f;
    Xs[i] = (bf16_t)v;
  }
  __syncthreads();
  {
    int ct = wave_id();  // 4 hidden tiles
    v8f a = tile_gemm(Xs, 128, Wt1t + ct * 16 * 128, 128, 128);
    store_lds_act(a, bt1, ct * 16, 64, Hs, 64, true);
  }
  __syncthreads();
  {
    int ct = wave_id();  // 4 output tiles
    v8f a = tile_gemm(Hs, 64, Wt2t + ct * 16 * 64, 64, 64);
    int n = ct * 16 + (lane_id() & 15), mo = (lane_id() >> 4) << 3;
    float bv = bt2[n];
#pragma unroll
    for (int r = 0; r < 8; ++r) {
      int row = row0 + r + mo, m = row & 63;
      out[(size_t)row * 64 + n] = (m < E_PER_G) ? a[r] + bv : 0.f;
    }
  }
}

// ---- logits[b] = (tq tc^T) / TEMP ------------------------------------------
__global__ void k_logits(const float* __restrict__ tq, const float* __restrict__ tc,
                         float* __restrict__ logits) {
  __shared__ __align__(16) bf16_t Qs[64 * 64], Cs[64 * 64];
  int b = blockIdx.x;
  for (int i = threadIdx.x; i < 4096; i += blockDim.x) {
    Qs[i] = (bf16_t)tq[(size_t)b * 4096 + i];
    Cs[i] = (bf16_t)tc[(size_t)b * 4096 + i];
  }
  __syncthreads();
  for (int job = wave_id(); job < 16; job += 8) {
    int mt = job >> 2, nt = job & 3;
    v8f a = tile_gemm(Qs + mt * 16 * 64, 64, Cs + nt * 16 * 64, 64, 64);
    int n = nt * 16 + (lane_id() & 15), mo = (lane_id() >> 4) << 3;
#pragma unroll
    for (int r = 0; r < 8; ++r)
      logits[(size_t)b * 4096 + (mt * 16 + r + mo) * 64 + n] = a[r] * 10.f;  // /TEMP
  }
}

// ---- Sinkhorn: 20 row/col logsumexp sweeps, then exp -----------------------
__global__ void k_sinkhorn(float* __restrict__ plan) {
  __shared__ float la[64 * 65];  // 65 stride -> conflict-free rows & cols
  int b = blockIdx.x, t = threadIdx.x;  // blockDim = 64
  for (int j = 0; j < 64; ++j) la[t * 65 + j] = plan[(size_t)b * 4096 + t * 64 + j];
  __syncthreads();
  for (int it = 0; it < 20; ++it) {
    float mx = -3.402823466e38f;
    for (int j = 0; j < 64; ++j) mx = fmaxf(mx, la[t * 65 + j]);
    float s = 0.f;
    for (int j = 0; j < 64; ++j) s += __expf(la[t * 65 + j] - mx);
    float lse = mx + __logf(s);
    for (int j = 0; j < 64; ++j) la[t * 65 + j] -= lse;
    __syncthreads();
    mx = -3.402823466e38f;
    for (int i = 0; i < 64; ++i) mx = fmaxf(mx, la[i * 65 + t]);
    s = 0.f;
    for (int i = 0; i < 64; ++i) s += __expf(la[i * 65 + t] - mx);
    lse = mx + __logf(s);
    for (int i = 0; i < 64; ++i) la[i * 65 + t] -= lse;
    __syncthreads();
  }
  for (int j = 0; j < 64; ++j) plan[(size_t)b * 4096 + t * 64 + j] = __expf(la[t * 65 + j]);
}

// ---- plan apply: interf(q) = plan @ C, interf(c) = plan^T @ Q --------------
__global__ void k_plan_apply(const float* __restrict__ plan, const float* __restrict__ msgs,
                             float* __restrict__ interf) {
  __shared__ __align__(16) bf16_t Pa[64 * 64], PaT[64 * 64];
  __shared__ __align__(16) bf16_t Ct[160 * 64], Qt[160 * 64];  // transposed chunks
  int b = blockIdx.x >> 2;
  int colbase = (blockIdx.x & 3) * 160;
  for (int i = threadIdx.x; i < 4096; i += blockDim.x) {
    float v = plan[(size_t)b * 4096 + i];
    Pa[i] = (bf16_t)v;
    PaT[(i & 63) * 64 + (i >> 6)] = (bf16_t)v;
  }
  for (int i = threadIdx.x; i < 160 * 64; i += blockDim.x) {
    int j = i >> 6, n = i & 63;
    int col = colbase + j;
    float vc = (n < E_PER_G) ? msgs[(size_t)((b * 2 + 1) * E_PER_G + n) * MSG_LD + col] : 0.f;
    float vq = (n < E_PER_G) ? msgs[(size_t)((b * 2 + 0) * E_PER_G + n) * MSG_LD + col] : 0.f;
    Ct[i] = (bf16_t)vc; Qt[i] = (bf16_t)vq;
  }
  __syncthreads();
  for (int job = wave_id(); job < 80; job += 8) {
    int sel = job / 40, rem = job % 40;
    int mt = rem / 10, ct = rem % 10;
    const bf16_t* A  = (sel ? PaT : Pa) + mt * 16 * 64;
    const bf16_t* Bm = (sel ? Qt : Ct) + ct * 16 * 64;
    v8f a = tile_gemm(A, 64, Bm, 64, 64);
    int n = ct * 16 + (lane_id() & 15), mo = (lane_id() >> 4) << 3;
#pragma unroll
    for (int r = 0; r < 8; ++r) {
      int m = mt * 16 + r + mo;
      if (m < E_PER_G)
        interf[(size_t)((b * 2 + sel) * E_PER_G + m) * MSG_LD + colbase + n] = a[r];
    }
  }
}

// ---- scores[b] = -sum relu(qf - plan @ cf) ---------------------------------
__global__ void k_scores(const float* __restrict__ plan, const float* __restrict__ msgs,
                         float* __restrict__ out) {
  __shared__ __align__(16) bf16_t Pa[64 * 64], CfT[128 * 64];
  __shared__ float red[256];
  int b = blockIdx.x;
  for (int i = threadIdx.x; i < 4096; i += blockDim.x)
    Pa[i] = (bf16_t)plan[(size_t)b * 4096 + i];
  for (int i = threadIdx.x; i < 128 * 64; i += blockDim.x) {
    int d = i >> 6, n = i & 63;
    CfT[i] = (bf16_t)((n < E_PER_G)
                 ? msgs[(size_t)((b * 2 + 1) * E_PER_G + n) * MSG_LD + 512 + d] : 0.f);
  }
  __syncthreads();
  float psum = 0.f;
  for (int job = wave_id(); job < 32; job += 8) {
    int mt = job >> 3, ct = job & 7;
    v8f a = tile_gemm(Pa + mt * 16 * 64, 64, CfT + ct * 16 * 64, 64, 64);
    int d = ct * 16 + (lane_id() & 15), mo = (lane_id() >> 4) << 3;
#pragma unroll
    for (int r = 0; r < 8; ++r) {
      int m = mt * 16 + r + mo;
      float qv = (m < E_PER_G)
                   ? msgs[(size_t)((b * 2) * E_PER_G + m) * MSG_LD + 512 + d] : 0.f;
      float v = qv - a[r];
      if (v > 0.f) psum += v;
    }
  }
  red[threadIdx.x] = psum;
  __syncthreads();
  for (int s = 128; s > 0; s >>= 1) {
    if ((int)threadIdx.x < s) red[threadIdx.x] += red[threadIdx.x + s];
    __syncthreads();
  }
  if (threadIdx.x == 0) out[b] = -red[0];
}

// ---------------------------------------------------------------------------
extern "C" void kernel_launch(void* const* d_in, const int* in_sizes, int n_in,
                              void* d_out, int out_size, void* d_ws, size_t ws_size,
                              hipStream_t stream) {
  const float* node_features = (const float*)d_in[0];
  const float* edge_features = (const float*)d_in[1];
  const int*   from_idx      = (const int*)d_in[2];
  const int*   to_idx        = (const int*)d_in[3];
  const float* W_ne = (const float*)d_in[4];  const float* b_ne = (const float*)d_in[5];
  const float* W_ee = (const float*)d_in[6];  const float* b_ee = (const float*)d_in[7];
  const float* Wc1  = (const float*)d_in[8];  const float* bc1  = (const float*)d_in[9];
  const float* Wc2  = (const float*)d_in[10]; const float* bc2  = (const float*)d_in[11];
  const float* Wm1  = (const float*)d_in[12]; const float* bm1  = (const float*)d_in[13];
  const float* Wm2  = (const float*)d_in[14]; const float* bm2  = (const float*)d_in[15];
  const float* Wr1  = (const float*)d_in[16]; const float* br1  = (const float*)d_in[17];
  const float* Wr2  = (const float*)d_in[18]; const float* br2  = (const float*)d_in[19];
  const float* Wn1  = (const float*)d_in[20]; const float* bn1  = (const float*)d_in[21];
  const float* Wn2  = (const float*)d_in[22]; const float* bn2  = (const float*)d_in[23];
  const float* Wt1  = (const float*)d_in[24]; const float* bt1  = (const float*)d_in[25];
  const float* Wt2  = (const float*)d_in[26]; const float* bt2  = (const float*)d_in[27];
  float* scores = (float*)d_out;

  // ---- workspace carve (bf16 weights first, then f32 buffers) ----
  char* ws = (char*)d_ws;
  size_t off = 0;
  auto abf = [&](size_t elems) { bf16_t* p = (bf16_t*)(ws + off);
                                 off = (off + elems * 2 + 15) & ~(size_t)15; return p; };
  bf16_t* Wne_t = abf(64 * 32);
  bf16_t* Wee_t = abf(32 * 32);
  bf16_t* Wc1t  = abf(160 * 160);
  bf16_t* Wc2t  = abf(32 * 160);
  bf16_t* Wm1t  = abf(128 * 160);
  bf16_t* Wm2t  = abf(128 * 128);
  bf16_t* Wr1t  = abf(128 * 160);
  bf16_t* Wr2t  = abf(128 * 128);
  bf16_t* Wn1t  = abf(128 * 192);
  bf16_t* Wn2t  = abf(64 * 128);
  bf16_t* Wt1t  = abf(64 * 128);
  bf16_t* Wt2t  = abf(64 * 64);
  auto af32 = [&](size_t elems) { float* p = (float*)(ws + off);
                                  off = (off + elems * 4 + 15) & ~(size_t)15; return p; };
  float* enc_n  = af32((size_t)NUM_NODES * 64);
  float* enc_e  = af32((size_t)NUM_EDGES * 32);
  float* nodes  = af32((size_t)NUM_NODES * 64);
  float* agg    = af32((size_t)NUM_NODES * 128);
  float* comb   = af32((size_t)NUM_EDGES * 32);
  float* msgs   = af32((size_t)NUM_EDGES * MSG_LD);
  float* interf = af32((size_t)NUM_EDGES * MSG_LD);
  float* tq     = af32((size_t)BATCH * 64 * 64);
  float* tc     = af32((size_t)BATCH * 64 * 64);
  float* plan   = af32((size_t)BATCH * 64 * 64);
  (void)ws_size; (void)n_in; (void)in_sizes; (void)out_size;

  auto cvt = [&](const float* W, bf16_t* Wt, int K, int N, int Kp, int Np) {
    int tot = Kp * Np;
    k_wcvt<<<(tot + 255) / 256, 256, 0, stream>>>(W, Wt, K, N, Kp, Np);
  };
  cvt(W_ne, Wne_t, 16, 64, 32, 64);
  cvt(W_ee, Wee_t, 16, 32, 32, 32);
  cvt(Wc1,  Wc1t, 160, 160, 160, 160);
  cvt(Wc2,  Wc2t, 160, 30, 160, 32);
  cvt(Wm1,  Wm1t, 158, 128, 160, 128);
  cvt(Wm2,  Wm2t, 128, 128, 128, 128);
  cvt(Wr1,  Wr1t, 158, 128, 160, 128);
  cvt(Wr2,  Wr2t, 128, 128, 128, 128);
  cvt(Wn1,  Wn1t, 192, 128, 192, 128);
  cvt(Wn2,  Wn2t, 128, 64, 128, 64);
  cvt(Wt1,  Wt1t, 128, 64, 128, 64);
  cvt(Wt2,  Wt2t, 64, 64, 64, 64);

  // encoders
  k_gemm_k32<<<NUM_NODES / 16, 128, 0, stream>>>(node_features, 16, 16, Wne_t, b_ne, 64, enc_n, 64);
  k_gemm_k32<<<NUM_EDGES / 16, 128, 0, stream>>>(edge_features, 16, 16, Wee_t, b_ee, 32, enc_e, 32);

  // interf (store[:, D:]) starts at zero
  {
    size_t n = (size_t)NUM_EDGES * MSG_LD;
    k_zero<<<(unsigned)((n + 255) / 256), 256, 0, stream>>>(interf, n);
  }

  for (int t = 0; t < 3; ++t) {
    { size_t n = (size_t)NUM_NODES * 64;
      k_copy<<<(unsigned)((n + 255) / 256), 256, 0, stream>>>(nodes, enc_n, n); }
    for (int p = 1; p <= 5; ++p) {
      k_comb<<<NUM_EDGES / 16, 128, 0, stream>>>(enc_e, interf,
          (p >= 2) ? (p - 2) * 128 : 0, (p >= 2) ? 1 : 0, Wc1t, bc1, Wc2t, bc2, comb);
      { size_t n = (size_t)NUM_NODES * 128;
        k_zero<<<(unsigned)((n + 255) / 256), 256, 0, stream>>>(agg, n); }
      k_msg<<<NUM_EDGES / 16, 128, 0, stream>>>(nodes, from_idx, to_idx, comb,
          Wm1t, bm1, Wm2t, bm2, Wr1t, br1, Wr2t, br2, msgs, (p - 1) * 128, agg);
      k_node<<<NUM_NODES / 16, 128, 0, stream>>>(nodes, agg, Wn1t, bn1, Wn2t, bn2);
    }
    k_tform<<<(BATCH * 64) / 16, 128, 0, stream>>>(msgs, 0, Wt1t, bt1, Wt2t, bt2, tq);
    k_tform<<<(BATCH * 64) / 16, 128, 0, stream>>>(msgs, 1, Wt1t, bt1, Wt2t, bt2, tc);
    k_logits<<<BATCH, 256, 0, stream>>>(tq, tc, plan);
    k_sinkhorn<<<BATCH, 64, 0, stream>>>(plan);
    if (t < 2)  // the final round's interaction features are never consumed
      k_plan_apply<<<BATCH * 4, 256, 0, stream>>>(plan, msgs, interf);
  }
  k_scores<<<BATCH, 256, 0, stream>>>(plan, msgs, scores);
}
